// OctonionLinear_43224550867287
// MI455X (gfx1250) — compile-verified, matrix operand order checked by reading, not run on hardware
//
#include <hip/hip_runtime.h>

typedef __attribute__((ext_vector_type(16))) __bf16 v16bf;
typedef __attribute__((ext_vector_type(8)))  __bf16 v8bf;
typedef __attribute__((ext_vector_type(8)))  float  v8f;

// Octonion sign table: e_a * e_b = OCT_S[a][b] * e_{a XOR b}
__constant__ float OCT_S[64] = {
  +1,+1,+1,+1,+1,+1,+1,+1,
  +1,-1,+1,-1,+1,-1,-1,+1,
  +1,-1,-1,+1,+1,+1,-1,-1,
  +1,+1,-1,-1,+1,-1,+1,-1,
  +1,-1,-1,-1,-1,+1,+1,+1,
  +1,+1,-1,+1,-1,-1,-1,+1,
  +1,+1,+1,-1,-1,+1,-1,-1,
  +1,-1,+1,+1,-1,-1,+1,-1,
};

// ---------------------------------------------------------------------------
// Kernel 1: expand octonion weight [O, I, 8] -> dense bf16 w_eff[N, K]
//   w_eff[(o*8+k)][(i*8+b)] = sign(a,b) * w[o][i][a]   with k = a ^ b
// ---------------------------------------------------------------------------
__global__ void build_weff_kernel(const float* __restrict__ w,
                                  __bf16* __restrict__ weff,
                                  int in_oct, int K) {
  const int t = blockIdx.x * blockDim.x + threadIdx.x;
  const int o = t / in_oct;
  const int i = t - o * in_oct;

  const float* wp = w + ((size_t)(o * in_oct + i) << 3);
  float wa[8];
#pragma unroll
  for (int a = 0; a < 8; ++a) wa[a] = wp[a];

  float blk[8][8];
#pragma unroll
  for (int a = 0; a < 8; ++a)
#pragma unroll
    for (int b = 0; b < 8; ++b)
      blk[a ^ b][b] = OCT_S[a * 8 + b] * wa[a];

#pragma unroll
  for (int k = 0; k < 8; ++k) {
    v8bf row;
#pragma unroll
    for (int b = 0; b < 8; ++b) row[b] = (__bf16)blk[k][b];
    *(v8bf*)(weff + (size_t)(o * 8 + k) * K + (i << 3)) = row;
  }
}

// ---------------------------------------------------------------------------
// Kernel 2: y = x * w_eff^T + bias.  bf16 WMMA / f32 accum.
//   Block 128x256, K-tile 64, 8 wave32s; wave tile 64x64 (4x4 frags, 2 K-steps
//   => 32 WMMAs per iteration, halving barrier frequency vs BK=32).
//   B tile: async global->LDS (ASYNCcnt).  A tile: reg-staged f32->bf16.
//   Double-buffered LDS, 1 barrier per K iteration.
// ---------------------------------------------------------------------------
#define BM 128
#define BN 256
#define BK 64
#define KPAD 80                    // LDS row stride (bf16): 128B data + 32B pad
#define A_ELEMS (BM * KPAD)        // 10240 bf16 per A buffer
#define B_ELEMS (BN * KPAD)        // 20480 bf16 per B buffer

__device__ __forceinline__ void async_b128_to_lds(unsigned lds_byte, const void* g) {
  asm volatile("global_load_async_to_lds_b128 %0, %1, off"
               :: "v"(lds_byte), "v"(g) : "memory");
}
__device__ __forceinline__ void wait_asynccnt0() {
  asm volatile("s_wait_asynccnt 0" ::: "memory");
}

__device__ __forceinline__ v16bf cvt16(float4 f0, float4 f1, float4 f2, float4 f3) {
  v16bf av;
  av[0]  = (__bf16)f0.x; av[1]  = (__bf16)f0.y; av[2]  = (__bf16)f0.z; av[3]  = (__bf16)f0.w;
  av[4]  = (__bf16)f1.x; av[5]  = (__bf16)f1.y; av[6]  = (__bf16)f1.z; av[7]  = (__bf16)f1.w;
  av[8]  = (__bf16)f2.x; av[9]  = (__bf16)f2.y; av[10] = (__bf16)f2.z; av[11] = (__bf16)f2.w;
  av[12] = (__bf16)f3.x; av[13] = (__bf16)f3.y; av[14] = (__bf16)f3.z; av[15] = (__bf16)f3.w;
  return av;
}

__global__ __launch_bounds__(256)
void oct_gemm_kernel(const float* __restrict__ x,
                     const __bf16* __restrict__ weff,
                     const float* __restrict__ bias,
                     float* __restrict__ out,
                     int M, int N, int K) {
  // single shared array so raw LDS byte offsets (base 0) are well defined
  __shared__ alignas(32) __bf16 smem[2 * A_ELEMS + 2 * B_ELEMS];  // 120 KB

  const int n0 = blockIdx.x * BN;
  const int m0 = blockIdx.y * BM;
  const int tid  = threadIdx.x;
  const int wave = tid >> 5;
  const int lane = tid & 31;
  const int wm = wave >> 2;            // 0..1  -> 64-row slab
  const int wn = wave & 3;             // 0..3  -> 64-col slab
  const int lr = lane & 15;
  const int lkoff = (lane >> 4) << 4;  // K offset 0/16 for upper lane half

  // A staging: thread -> (row = tid>>1, 32-float half-row)
  const int arow = tid >> 1;
  const int aseg = (tid & 1) << 5;     // 0 or 32 (K elements)
  const float* xg = x + (size_t)(m0 + arow) * K + aseg;

  // B staging: thread -> full 128B row (8 x b128 async chunks)
  const __bf16* wgrow = weff + (size_t)(n0 + tid) * K;
  const unsigned blds0 = (unsigned)((2 * A_ELEMS + tid * KPAD) * 2);  // bytes

  auto issueB = [&](int buf, int kt) {
    const __bf16* g = wgrow + kt * BK;
    const unsigned l = blds0 + (unsigned)buf * (B_ELEMS * 2);
#pragma unroll
    for (int c = 0; c < 8; ++c)
      async_b128_to_lds(l + c * 16, g + c * 8);
  };

  v8f acc[4][4] = {};
  const int nk = K / BK;

  // ---- prologue: tile 0 staged, tile 1 A-loads in flight, tile 0 B async ----
  issueB(0, 0);
  {
    const float4* xp = (const float4*)xg;
    *(v16bf*)&smem[arow * KPAD + aseg]      = cvt16(xp[0], xp[1], xp[2], xp[3]);
    *(v16bf*)&smem[arow * KPAD + aseg + 16] = cvt16(xp[4], xp[5], xp[6], xp[7]);
  }
  float4 f[8];
  if (nk > 1) {
    const float4* xp = (const float4*)(xg + BK);
#pragma unroll
    for (int c = 0; c < 8; ++c) f[c] = xp[c];
  }

  for (int t = 0; t < nk; ++t) {
    const int cur = t & 1, nxt = cur ^ 1;

    wait_asynccnt0();        // B(t) landed in LDS
    __syncthreads();         // all waves done with buffers being overwritten

    if (t + 1 < nk) {
      issueB(nxt, t + 1);                                  // async B(t+1)
      __bf16* as = &smem[nxt * A_ELEMS + arow * KPAD + aseg];
      *(v16bf*)as        = cvt16(f[0], f[1], f[2], f[3]);  // A(t+1) regs->LDS
      *(v16bf*)(as + 16) = cvt16(f[4], f[5], f[6], f[7]);
      if (t + 2 < nk) {                                    // A(t+2) into regs
        const float4* xp = (const float4*)(xg + (t + 2) * BK);
#pragma unroll
        for (int c = 0; c < 8; ++c) f[c] = xp[c];
      }
    }

    // ---- compute tile t: 2 K-steps x 16 WMMAs per wave ----
    const __bf16* As = &smem[cur * A_ELEMS];
    const __bf16* Bs = &smem[2 * A_ELEMS + cur * B_ELEMS];
#pragma unroll
    for (int s = 0; s < 2; ++s) {
      const int koff = s * 32 + lkoff;
      v16bf afr[4], bfr[4];
#pragma unroll
      for (int mt = 0; mt < 4; ++mt)
        afr[mt] = *(const v16bf*)&As[(wm * 64 + mt * 16 + lr) * KPAD + koff];
#pragma unroll
      for (int nt = 0; nt < 4; ++nt)
        bfr[nt] = *(const v16bf*)&Bs[(wn * 64 + nt * 16 + lr) * KPAD + koff];

#pragma unroll
      for (int mt = 0; mt < 4; ++mt)
#pragma unroll
        for (int nt = 0; nt < 4; ++nt)
          acc[mt][nt] = __builtin_amdgcn_wmma_f32_16x16x32_bf16(
              false, afr[mt], false, bfr[nt],
              (short)0, acc[mt][nt], false, false);
    }
  }

  // ---- epilogue: VGPR r -> lanes0-15: M=r, lanes16-31: M=r+8 ----
  const int mrowbase = m0 + wm * 64 + ((lane >> 4) << 3);
  const int colbase  = n0 + wn * 64 + (lane & 15);
#pragma unroll
  for (int nt = 0; nt < 4; ++nt) {
    const int col = colbase + nt * 16;
    const float bvadd = bias[col];
#pragma unroll
    for (int mt = 0; mt < 4; ++mt) {
      float* op = out + (size_t)(mrowbase + mt * 16) * N + col;
#pragma unroll
      for (int r = 0; r < 8; ++r)
        op[(size_t)r * N] = acc[mt][nt][r] + bvadd;
    }
  }
}

// ---------------------------------------------------------------------------
extern "C" void kernel_launch(void* const* d_in, const int* in_sizes, int n_in,
                              void* d_out, int out_size, void* d_ws, size_t ws_size,
                              hipStream_t stream) {
  const float* x    = (const float*)d_in[0];
  const float* w    = (const float*)d_in[1];
  const float* bias = (const float*)d_in[2];
  float* out = (float*)d_out;
  __bf16* weff = (__bf16*)d_ws;   // [N, K] bf16 = 32 MB

  const int N = in_sizes[2];                              // 4096
  const int K = (int)(((long long)in_sizes[1] * 8) / N);  // 4096
  const int M = in_sizes[0] / K;                          // 16384
  const int out_oct = N >> 3, in_oct = K >> 3;

  build_weff_kernel<<<(out_oct * in_oct) / 256, 256, 0, stream>>>(w, weff, in_oct, K);

  dim3 grid(N / BN, M / BM);
  oct_gemm_kernel<<<grid, 256, 0, stream>>>(x, weff, bias, out, M, N, K);
}
